// MVKE_28673201668372
// MI455X (gfx1250) — matrix operand order repaired and checked by reading, not compiled, not run on hardware
//
#include <hip/hip_runtime.h>

// ---------------- problem constants ----------------
constexpr int cE = 100;   // embedding dim
constexpr int cV = 3;     // virtual kernels
constexpr int cS = 100;   // sequence
constexpr int cT = 10;    // tags
constexpr int KP = 128;   // padded K pitch (halves): 4 k-tiles of 32
constexpr int MP = 112;   // padded M/N rows (7 tiles of 16)
constexpr int P1 = 105;   // odd f32 pitch (bank-conflict-free)
constexpr float SCL  = 0.1f;              // 1/sqrt(100)
constexpr float PADV = -4294967295.0f;    // -(2^32)+1, rounds to -2^32 in f32 like the ref

typedef __attribute__((ext_vector_type(16))) _Float16 v16h;
typedef __attribute__((ext_vector_type(8)))  _Float16 v8h;
typedef __attribute__((ext_vector_type(8)))  float    v8f;

__device__ __forceinline__ v16h cat8(v8h a, v8h b) {
  return __builtin_shufflevector(a, b, 0,1,2,3,4,5,6,7,8,9,10,11,12,13,14,15);
}

__device__ __forceinline__ void store4h(_Float16* p, float a, float b, float c, float d) {
  union { _Float16 h[4]; uint2 u; } pk;
  pk.h[0] = (_Float16)a; pk.h[1] = (_Float16)b; pk.h[2] = (_Float16)c; pk.h[3] = (_Float16)d;
  *(uint2*)p = pk.u;
}

// ---------------- prep kernel: f16 padded weights, of2/of3 ----------------
__global__ void __launch_bounds__(256) mvke_prep(
    const float* __restrict__ fc1w, const float* __restrict__ fc4w,
    const float* __restrict__ vk,
    const float* __restrict__ fc2w, const float* __restrict__ fc2b,
    const float* __restrict__ fc3w, const float* __restrict__ fc3b,
    _Float16* __restrict__ w1h, _Float16* __restrict__ w4h,
    float* __restrict__ of2, float* __restrict__ of3)
{
  const int tid = threadIdx.x + blockIdx.x * blockDim.x;
  const int nthr = blockDim.x * gridDim.x;
  for (int i = tid; i < MP * KP; i += nthr) {
    int n = i / KP, k = i % KP;
    float a = (n < cE && k < cE) ? fc1w[n * cE + k] : 0.0f;
    float c = (n < cE && k < cE) ? fc4w[n * cE + k] : 0.0f;
    w1h[i] = (_Float16)a;
    w4h[i] = (_Float16)c;
  }
  for (int i = tid; i < cV * KP; i += nthr) {
    int v = i / KP, e = i % KP;
    float a2 = 0.0f, a3 = 0.0f;
    if (e < cE) {
      for (int k = 0; k < cE; ++k) {
        float q = vk[v * cE + k];
        a2 += q * fc2w[e * cE + k];
        a3 += q * fc3w[e * cE + k];
      }
      a2 += fc2b[e]; a3 += fc3b[e];
    }
    of2[i] = a2; of3[i] = a3;
  }
}

// ---------------- main fused kernel: one block per batch element ----------------
struct LdsLayout {
  alignas(16) _Float16 x16 [MP * KP];   // x[b] f16 (A of fc1), zero-padded
  alignas(16) _Float16 w1  [MP * KP];   // fc1 weights f16 (row n, k contiguous)
  alignas(16) _Float16 w4  [MP * KP];   // fc4 weights f16
  alignas(16) _Float16 of1T[MP * KP];   // of1 transposed f16: [e][s]
  alignas(16) _Float16 sm1h[16 * KP];   // sm1 f16: [v][s], rows 3..15 zero
  alignas(16) _Float16 tg16[16 * KP];   // tag[b] f16 (A of fc4), zero-padded
  alignas(16) float of4 [cT * P1];      // stage-2 activations f32
  alignas(16) float tg32[cT * 104];     // tag[b] f32 (final dot)
  alignas(16) float of2 [cV * KP];
  alignas(16) float of3 [cV * KP];
  alignas(16) float b1[MP];
  alignas(16) float b4[MP];
  alignas(16) float sgn[cS];
  alignas(16) float ws1[cV * 104];      // weighted summaries f32 [v][e]
  alignas(16) float m2 [cV * 16];
  alignas(16) float sm2[cV * 16];
  alignas(16) float dvt[cV * 16];
};

__global__ void __launch_bounds__(256) mvke_main(
    const float* __restrict__ x, const float* __restrict__ tag,
    const _Float16* __restrict__ w1h, const _Float16* __restrict__ w4h,
    const float* __restrict__ of2g, const float* __restrict__ of3g,
    const float* __restrict__ b1g, const float* __restrict__ b4g,
    float* __restrict__ out)
{
  __shared__ LdsLayout L;
  const int tid  = threadIdx.x;
  const int wv   = tid >> 5;
  const int lane = tid & 31;
  const int b    = blockIdx.x;

  // fragment lane constants (contiguous-K A/B, wave32)
  const int l15  = lane & 15;
  const int asel = (lane & 16) ? 16 : 0;   // A frag: lanes>=16 hold K+16
  const int bsel = (lane & 16) ? 16 : 0;   // B frag: lanes>=16 hold K+16
  const int msel = (lane & 16) ? 8  : 0;   // D frag: lanes>=16 hold M+8

  // ---------- phase 0: stage LDS ----------
  for (int i = tid; i < (MP * KP) / 8; i += 256) {   // weights: 16B copies
    ((uint4*)L.w1)[i] = ((const uint4*)w1h)[i];
    ((uint4*)L.w4)[i] = ((const uint4*)w4h)[i];
  }
  for (int i = tid; i < 12 * KP; i += 256) L.x16[100 * KP + i] = (_Float16)0.0f;  // A rows 100..111
  for (int i = tid; i < 6  * KP; i += 256) L.tg16[10 * KP + i] = (_Float16)0.0f;  // A rows 10..15
  for (int i = tid; i < 13 * KP; i += 256) L.sm1h[3 * KP + i] = (_Float16)0.0f;   // sm1h rows 3..15
  for (int i = tid; i < MP * 16; i += 256) {                                      // of1T K-pad cols 112..127
    L.of1T[(i >> 4) * KP + 112 + (i & 15)] = (_Float16)0.0f;
  }
  for (int i = tid; i < cV * KP; i += 256) { L.of2[i] = of2g[i]; L.of3[i] = of3g[i]; }
  for (int i = tid; i < MP; i += 256) {
    L.b1[i] = (i < cE) ? b1g[i] : 0.0f;
    L.b4[i] = (i < cE) ? b4g[i] : 0.0f;
  }

  // x[b]: wave per row; lane<25 loads float4 (coalesced); |x| row-sum for mask
  const float* xb = x + (size_t)b * cS * cE;
  for (int s = wv; s < cS; s += 8) {
    float4 v = make_float4(0.f, 0.f, 0.f, 0.f);
    if (lane < 25) v = ((const float4*)(xb + (size_t)s * cE))[lane];
    float asum = fabsf(v.x) + fabsf(v.y) + fabsf(v.z) + fabsf(v.w);
    for (int off = 16; off; off >>= 1) asum += __shfl_xor(asum, off);
    if (lane == 0) L.sgn[s] = asum;
    if (lane < 25) store4h(&L.x16[s * KP + lane * 4], v.x, v.y, v.z, v.w);
    else           store4h(&L.x16[s * KP + 100 + (lane - 25) * 4], 0.f, 0.f, 0.f, 0.f);
  }
  // tag[b]: f32 copy + f16 tile
  const float* tb = tag + (size_t)b * cT * cE;
  for (int t = wv; t < cT; t += 8) {
    float4 v = make_float4(0.f, 0.f, 0.f, 0.f);
    if (lane < 25) {
      v = ((const float4*)(tb + (size_t)t * cE))[lane];
      *(float4*)&L.tg32[t * 104 + lane * 4] = v;
      store4h(&L.tg16[t * KP + lane * 4], v.x, v.y, v.z, v.w);
    } else {
      store4h(&L.tg16[t * KP + 100 + (lane - 25) * 4], 0.f, 0.f, 0.f, 0.f);
    }
  }
  __syncthreads();

  // ---------- phase 1: WMMA GEMMs ----------
  // fc1: of1 = x[b] @ W1^T + b1, stored transposed as of1T[e][s] (1 ds_store_b128/lane/tile)
  for (int tile = wv; tile < 49; tile += 8) {
    const int mt = tile / 7, nt = tile % 7;
    v8f acc = {};
    const int arow = mt * 16 + l15;
    const int brow = nt * 16 + l15;
#pragma unroll
    for (int kt = 0; kt < 4; ++kt) {
      const v8h* ap = (const v8h*)&L.x16[arow * KP + kt * 32 + asel];
      v16h a = cat8(ap[0], ap[1]);                 // 16 contiguous K
      const v8h* bp = (const v8h*)&L.w1[brow * KP + kt * 32 + bsel];
      v16h bm = cat8(bp[0], bp[1]);                // 16 contiguous K
      acc = __builtin_amdgcn_wmma_f32_16x16x32_f16(false, a, false, bm,
                                                   (short)0, acc, false, false);
    }
    const int n     = nt * 16 + l15;     // e index (column of of1)
    const int mbase = mt * 16 + msel;    // s base (rows of of1)
    const float bias = L.b1[n];          // zero for n>=100 (w1 rows zeroed too)
    union { _Float16 h[8]; uint4 u; } pk;
#pragma unroll
    for (int r = 0; r < 8; ++r) {
      float vv = acc[r] + bias;
      if (mbase + r >= cS) vv = 0.0f;    // keep K-pad (s>=100) exactly zero
      pk.h[r] = (_Float16)vv;
    }
    *(uint4*)&L.of1T[n * KP + mbase] = pk.u;       // transposed, contiguous in s
  }
  // fc4: of4 = tag[b] @ W4^T + b4  (1x7 tiles, waves 0..6)
  if (wv < 7) {
    const int nt = wv;
    v8f acc = {};
    const int brow = nt * 16 + l15;
#pragma unroll
    for (int kt = 0; kt < 4; ++kt) {
      const v8h* ap = (const v8h*)&L.tg16[l15 * KP + kt * 32 + asel];
      v16h a = cat8(ap[0], ap[1]);
      const v8h* bp = (const v8h*)&L.w4[brow * KP + kt * 32 + bsel];
      v16h bm = cat8(bp[0], bp[1]);
      acc = __builtin_amdgcn_wmma_f32_16x16x32_f16(false, a, false, bm,
                                                   (short)0, acc, false, false);
    }
    const int n = nt * 16 + l15;
    if (n < cE) {
      const float bias = L.b4[n];
      if (lane < 16) {                       // rows 0..7 all valid (<10)
#pragma unroll
        for (int r = 0; r < 8; ++r) L.of4[r * P1 + n] = acc[r] + bias;
      } else {                               // rows 8..15: only 8,9 valid
        L.of4[8 * P1 + n] = acc[0] + bias;
        L.of4[9 * P1 + n] = acc[1] + bias;
      }
    }
  }
  __syncthreads();

  // ---------- phase 2 (fused): m1 + S-softmax in registers (waves 0..2); m2 + T-softmax (wave 3) ----------
  if (wv < cV) {
    const int v  = wv;
    const int s0 = lane * 4;                 // 4 consecutive s per lane; 128 slots total
    float a0 = 0.f, a1 = 0.f, a2 = 0.f, a3 = 0.f;
    for (int e = 0; e < cE; ++e) {
      const float w = L.of2[v * KP + e];     // broadcast
      union { uint2 u; _Float16 h[4]; } pk;
      pk.u = *(const uint2*)&L.of1T[e * KP + s0];   // 64 banks, conflict-free
      a0 += (float)pk.h[0] * w;
      a1 += (float)pk.h[1] * w;
      a2 += (float)pk.h[2] * w;
      a3 += (float)pk.h[3] * w;
    }
    float lg[4] = { a0 * SCL, a1 * SCL, a2 * SCL, a3 * SCL };
#pragma unroll
    for (int j = 0; j < 4; ++j) {
      const int s = s0 + j;
      if (s >= cS)                lg[j] = -3.0e38f;   // fake slot: exp -> 0 always
      else if (L.sgn[s] == 0.0f)  lg[j] = PADV;       // masked behavior
    }
    float mx = fmaxf(fmaxf(lg[0], lg[1]), fmaxf(lg[2], lg[3]));
    for (int off = 16; off; off >>= 1) mx = fmaxf(mx, __shfl_xor(mx, off));
    float ev[4], sum = 0.f;
#pragma unroll
    for (int j = 0; j < 4; ++j) { ev[j] = __expf(lg[j] - mx); sum += ev[j]; }
    for (int off = 16; off; off >>= 1) sum += __shfl_xor(sum, off);
    const float inv = 1.0f / sum;
    union { _Float16 h[4]; uint2 u; } o4;
#pragma unroll
    for (int j = 0; j < 4; ++j) o4.h[j] = (_Float16)(ev[j] * inv);
    *(uint2*)&L.sm1h[v * KP + s0] = o4.u;    // writes zeros into K-pad cols too
  }
  if (wv == 3) {
    if (lane < cV * cT) {
      const int v = lane / cT, t = lane % cT;
      float acc = 0.f;
      for (int e = 0; e < cE; ++e) acc += L.of4[t * P1 + e] * L.of3[v * KP + e];
      L.m2[v * 16 + t] = acc * SCL;
    }
    // same wave: LDS ops are in-order, so lanes 0..2 can read m2 written above
    if (lane < cV) {
      const int v = lane;
      float mx = -3.0e38f;
      for (int t = 0; t < cT; ++t) mx = fmaxf(mx, L.m2[v * 16 + t]);
      float sum = 0.f;
      for (int t = 0; t < cT; ++t) {
        float e2 = __expf(L.m2[v * 16 + t] - mx);
        L.sm2[v * 16 + t] = e2;
        sum += e2;
      }
      const float inv = 1.0f / sum;
      for (int t = 0; t < cT; ++t) L.sm2[v * 16 + t] *= inv;
    }
  }
  __syncthreads();

  // ---------- phase 3: ws1 via WMMA: ws1T[e,v] = sum_s of1T[e,s] * sm1[s,v] ----------
  // A = of1T (M=e, K=s), B = sm1h (rows = v, K=s contiguous), 7 M-tiles on waves 0..6
  if (wv < 7) {
    const int et = wv;
    v8f acc = {};
    const int arow = et * 16 + l15;
#pragma unroll
    for (int kt = 0; kt < 4; ++kt) {
      const v8h* ap = (const v8h*)&L.of1T[arow * KP + kt * 32 + asel];
      v16h a = cat8(ap[0], ap[1]);
      const v8h* bp = (const v8h*)&L.sm1h[l15 * KP + kt * 32 + bsel];
      v16h bm = cat8(bp[0], bp[1]);
      acc = __builtin_amdgcn_wmma_f32_16x16x32_f16(false, a, false, bm,
                                                   (short)0, acc, false, false);
    }
    if (l15 < cV) {                      // D: lane col = v, rows = e
      const int eb = et * 16 + msel;
#pragma unroll
      for (int r = 0; r < 8; ++r) {
        int e = eb + r;
        if (e < cE) L.ws1[l15 * 104 + e] = acc[r];
      }
    }
  }
  __syncthreads();

  // ---------- phase 4: dvt[v,t] = <ws1[v], tag[t]> (float4 LDS loads) ----------
  for (int i = tid; i < cV * cT; i += 256) {
    const int v = i / cT, t = i % cT;
    float acc = 0.f;
    for (int e = 0; e < cE; e += 4) {
      const float4 a4 = *(const float4*)&L.ws1[v * 104 + e];
      const float4 c4 = *(const float4*)&L.tg32[t * 104 + e];
      acc += a4.x * c4.x + a4.y * c4.y + a4.z * c4.z + a4.w * c4.w;
    }
    L.dvt[v * 16 + t] = acc;
  }
  __syncthreads();

  // ---------- phase 5: out[b,t] = sum_v sm2[t,v] * dvt[v,t] ----------
  if (tid < cT) {
    float acc = 0.f;
    for (int v = 0; v < cV; ++v) acc += L.sm2[v * 16 + tid] * L.dvt[v * 16 + tid];
    out[(size_t)b * cT + tid] = acc;
  }
}

// ---------------- host launcher ----------------
extern "C" void kernel_launch(void* const* d_in, const int* in_sizes, int n_in,
                              void* d_out, int out_size, void* d_ws, size_t ws_size,
                              hipStream_t stream) {
  (void)in_sizes; (void)n_in; (void)out_size; (void)ws_size;
  const float* x    = (const float*)d_in[0];
  const float* tag  = (const float*)d_in[1];
  const float* vk   = (const float*)d_in[2];
  const float* fc1w = (const float*)d_in[3];
  const float* fc1b = (const float*)d_in[4];
  const float* fc2w = (const float*)d_in[5];
  const float* fc2b = (const float*)d_in[6];
  const float* fc3w = (const float*)d_in[7];
  const float* fc3b = (const float*)d_in[8];
  const float* fc4w = (const float*)d_in[9];
  const float* fc4b = (const float*)d_in[10];

  char* ws = (char*)d_ws;
  _Float16* w1h = (_Float16*)(ws);                 // 112*128 halves = 28672 B
  _Float16* w4h = (_Float16*)(ws + 28672);         // 28672 B
  float*    of2 = (float*)(ws + 57344);            // 3*128 f32 = 1536 B
  float*    of3 = (float*)(ws + 57344 + 1536);     // 1536 B

  mvke_prep<<<16, 256, 0, stream>>>(fc1w, fc4w, vk, fc2w, fc2b, fc3w, fc3b,
                                    w1h, w4h, of2, of3);
  mvke_main<<<8192, 256, 0, stream>>>(x, tag, w1h, w4h, of2, of3, fc1b, fc4b,
                                      (float*)d_out);
}
